// QuasarAttention_32177894981754
// MI455X (gfx1250) — matrix operand order repaired and checked by reading, not compiled, not run on hardware
//
#include <hip/hip_runtime.h>
#include <cstdint>

// Problem constants (match reference)
#define BB   2
#define TT   2048
#define HIDc 2048
#define HH   16
#define DD   128
#define KDIM 2048
#define CH   32            // scan chunk = WMMA K-step for bf16

typedef __attribute__((ext_vector_type(16))) __bf16 v16bf;
typedef __attribute__((ext_vector_type(8)))  float  v8f;
typedef __attribute__((ext_vector_type(8)))  unsigned short u16x8;

union FragU {
  v16bf v;
  u16x8 h[2];
  unsigned short s[16];
};

__device__ __forceinline__ unsigned short f2bf(float x) {
  unsigned u = __float_as_uint(x);
  u += 0x7FFFu + ((u >> 16) & 1u);           // round-to-nearest-even
  return (unsigned short)(u >> 16);
}
__device__ __forceinline__ float bf2f(unsigned short h) {
  return __uint_as_float(((unsigned)h) << 16);
}
__device__ __forceinline__ float sigmoidf_(float x) { return 1.f / (1.f + __expf(-x)); }

__device__ __forceinline__ v16bf ldfrag(const unsigned short* p0, const unsigned short* p1) {
  FragU f;
  f.h[0] = *(const u16x8*)p0;
  f.h[1] = *(const u16x8*)p1;
  return f.v;
}
__device__ __forceinline__ v8f wmma_bf16(v16bf a, v16bf b, v8f c) {
  // D(f32 16x16) = A(bf16 16x32) * B(bf16 32x16) + C
  return __builtin_amdgcn_wmma_f32_16x16x32_bf16(false, a, false, b, (short)0, c, false, false);
}
__device__ __forceinline__ v8f zero8() { v8f c; for (int i = 0; i < 8; ++i) c[i] = 0.f; return c; }

// CDNA5 async global->LDS copy (ASYNCcnt-tracked DMA, bypasses VGPRs).
// Generic LDS pointer's low 32 bits are the LDS byte address (ISA 10.2 aperture rule).
__device__ __forceinline__ void async_copy_b128(void* lds_dst, const void* gsrc) {
  asm volatile("global_load_async_to_lds_b128 %0, %1, off"
               :: "v"((unsigned)(uintptr_t)lds_dst), "v"(gsrc)
               : "memory");
}
__device__ __forceinline__ void wait_async0() {
  asm volatile("s_wait_asynccnt 0x0" ::: "memory");
}

// ---------------------------------------------------------------------------
// One-time f32 -> bf16 pack (elementwise).
// ---------------------------------------------------------------------------
__global__ __launch_bounds__(256) void pack_bf16_kernel(
    const float* __restrict__ src, unsigned short* __restrict__ dst, int n)
{
  for (int i = blockIdx.x * 256 + threadIdx.x; i < n; i += gridDim.x * 256)
    dst[i] = f2bf(src[i]);
}

// ---------------------------------------------------------------------------
// One-time transpose+convert: BT[n][k] = bf16(B[k][n]).  32x32 LDS tiles.
// ---------------------------------------------------------------------------
__global__ __launch_bounds__(256) void transpose_bf16_kernel(
    const float* __restrict__ B, unsigned short* __restrict__ BT, int K, int N)
{
  __shared__ unsigned short tile[32][33];
  const int tid = threadIdx.x;
  const int tx = tid & 31, ty = tid >> 5;            // 8 rows per pass
  const int k0 = blockIdx.y * 32, n0 = blockIdx.x * 32;
  #pragma unroll
  for (int p = 0; p < 4; ++p) {
    const int k = k0 + ty + p * 8, n = n0 + tx;
    tile[ty + p * 8][tx] = (k < K && n < N) ? f2bf(B[(size_t)k * N + n]) : (unsigned short)0;
  }
  __syncthreads();
  #pragma unroll
  for (int p = 0; p < 4; ++p) {
    const int n = n0 + ty + p * 8, k = k0 + tx;
    if (n < N && k < K) BT[(size_t)n * K + k] = tile[tx][ty + p * 8];
  }
}

// ---------------------------------------------------------------------------
// C = A[MxK](bf16) * BT[NxK](bf16)^T (+bias). f32 and/or bf16 outputs.
// 128x128 tile, K-step 32, double-buffered async global->LDS staging:
// one barrier per K-tile; tile i+1 DMA overlaps tile i's 8-WMMA burst.
// ---------------------------------------------------------------------------
__global__ __launch_bounds__(256) void gemm_bf16_kernel(
    const unsigned short* __restrict__ A,   // [M][K] bf16
    const unsigned short* __restrict__ BT,  // [N][K] bf16 (pre-transposed)
    const float* __restrict__ bias,
    float* __restrict__ C,                  // optional f32 out
    unsigned short* __restrict__ Cbf,       // optional bf16 out
    int M, int N, int K)
{
  __shared__ unsigned short aS[2][128][32];   // [m][k]
  __shared__ unsigned short bS[2][128][32];   // [n][k]
  const int tid  = threadIdx.x;
  const int wave = tid >> 5, lane = tid & 31, lo = lane & 15, hi = lane >> 4;
  const int row0 = blockIdx.y * 128, col0 = blockIdx.x * 128;

  v8f acc[8];
  for (int i = 0; i < 8; ++i) acc[i] = zero8();

  const int sr = tid >> 1;                // staging row 0..127
  const int sc = (tid & 1) * 16;          // staging k-offset 0/16
  const int arow = row0 + sr;
  const int bcol = col0 + sr;

  auto issue_tile = [&](int kb, int buf) {
    if (arow < M) {
      const unsigned short* ga = A + (size_t)arow * K + kb + sc;
      async_copy_b128(&aS[buf][sr][sc],     ga);
      async_copy_b128(&aS[buf][sr][sc + 8], ga + 8);
    }
    if (bcol < N) {
      const unsigned short* gb = BT + (size_t)bcol * K + kb + sc;
      async_copy_b128(&bS[buf][sr][sc],     gb);
      async_copy_b128(&bS[buf][sr][sc + 8], gb + 8);
    }
  };

  const int nK = K >> 5;
  issue_tile(0, 0);
  for (int it = 0; it < nK; ++it) {
    wait_async0();                          // this wave's tile `it` landed in LDS
    __syncthreads();                        // all waves' tile `it` visible; prev reads done
    if (it + 1 < nK) issue_tile((it + 1) << 5, (it + 1) & 1);
    const int buf = it & 1;
    v16bf af = ldfrag(&aS[buf][wave * 16 + lo][hi * 8], &aS[buf][wave * 16 + lo][16 + hi * 8]);
    #pragma unroll
    for (int nt = 0; nt < 8; ++nt) {
      v16bf bf = ldfrag(&bS[buf][nt * 16 + lo][hi * 8], &bS[buf][nt * 16 + lo][16 + hi * 8]);
      acc[nt] = wmma_bf16(af, bf, acc[nt]);
    }
  }
  #pragma unroll
  for (int nt = 0; nt < 8; ++nt) {
    const int cc = col0 + nt * 16 + lo;
    if (cc < N) {
      const float badd = bias ? bias[cc] : 0.f;
      #pragma unroll
      for (int j = 0; j < 8; ++j) {
        const int rr = row0 + wave * 16 + hi * 8 + j;
        if (rr < M) {
          const float v = acc[nt][j] + badd;
          if (C)   C[(size_t)rr * N + cc]   = v;
          if (Cbf) Cbf[(size_t)rr * N + cc] = f2bf(v);
        }
      }
    }
  }
}

// ---------------------------------------------------------------------------
// Causal depthwise conv (K=4) + SiLU (+ optional per-head l2norm).
// bf16 in / bf16 out (memory-bound: halve bytes against 23.3 TB/s roofline).
// One wave per (b,t,h); lane handles 4 contiguous channels of the head.
// ---------------------------------------------------------------------------
__global__ __launch_bounds__(256) void conv_silu_kernel(
    const unsigned short* __restrict__ pre, const float* __restrict__ wconv,
    unsigned short* __restrict__ out, int do_norm)
{
  const int tid  = threadIdx.x;
  const int wave = tid >> 5, lane = tid & 31;
  const int g = blockIdx.x * 8 + wave;           // 0 .. B*T*H-1
  const int h = g & (HH - 1);
  const int t = (g >> 4) & (TT - 1);
  const int b = g >> 15;
  const int c = h * DD + lane * 4;
  const size_t row = (size_t)b * TT + t;

  float acc[4] = {0.f, 0.f, 0.f, 0.f};
  #pragma unroll
  for (int j = 0; j < 4; ++j) {                  // y[t] = sum_j x[t-j]*w[3-j]
    if (t - j < 0) continue;
    const unsigned short* xr = pre + (row - (size_t)j) * KDIM + c;
    #pragma unroll
    for (int i = 0; i < 4; ++i) acc[i] += bf2f(xr[i]) * wconv[(c + i) * 4 + (3 - j)];
  }
  #pragma unroll
  for (int i = 0; i < 4; ++i) acc[i] = acc[i] * sigmoidf_(acc[i]);   // SiLU
  if (do_norm) {
    float ss = acc[0]*acc[0] + acc[1]*acc[1] + acc[2]*acc[2] + acc[3]*acc[3];
    for (int m = 16; m >= 1; m >>= 1) ss += __shfl_xor(ss, m, 32);   // wave32
    const float inv = 1.f / fmaxf(sqrtf(ss), 1e-12f);
    #pragma unroll
    for (int i = 0; i < 4; ++i) acc[i] *= inv;
  }
  unsigned short* orow = out + row * KDIM + c;
  #pragma unroll
  for (int i = 0; i < 4; ++i) orow[i] = f2bf(acc[i]);
}

// ---------------------------------------------------------------------------
// Chunked gated linear-attention scan. One workgroup (8 waves) per (b,h).
// Chunk C=32: intra-chunk attention (32x32) with per-channel decay (safe
// factorization: only decaying exponents), rank-32 state update, all heavy
// math via bf16 WMMA. State S[128][128] f32 in LDS; wave w exclusively owns
// value-columns dv = w*16..w*16+15 of S and O (no barrier inside phase 3).
// ---------------------------------------------------------------------------
__global__ __launch_bounds__(256) void quasar_scan_kernel(
    const unsigned short* __restrict__ q, const unsigned short* __restrict__ k,
    const unsigned short* __restrict__ v, const float* __restrict__ beta_logit,
    const float* __restrict__ A_log, const float* __restrict__ dt_bias,
    float* __restrict__ o)
{
  __shared__ float S[DD][DD];                    // state, f32            64 KB
  __shared__ float lampow[CH + 1][DD];           // lambda^e, e=0..32     16.5 KB
  __shared__ float wdec[DD];
  __shared__ unsigned short qb[CH][DD];          // raw q (bf16)
  __shared__ unsigned short kb_[CH][DD];         // raw k (bf16)
  __shared__ unsigned short qsb[CH][DD];         // q * lam^{r+1}
  __shared__ unsigned short kdbT[DD][CH];        // (k * lam^{31-r})^T
  __shared__ unsigned short vbbT[DD][CH];        // (beta * v)^T
  __shared__ unsigned short qlb[16][DD];         // q_{16+m} * lam^{m}
  __shared__ unsigned short klb[16][DD];         // k_{s}    * lam^{16-s}
  __shared__ unsigned short Ab[CH][CH];          // intra attention (bf16)

  const int tid  = threadIdx.x;
  const int wave = tid >> 5, lane = tid & 31, lo = lane & 15, hi = lane >> 4;
  const int b = blockIdx.x >> 4, h = blockIdx.x & 15;

  if (tid < DD) {                                // per-channel decay rate
    const float aexp = __expf(A_log[h]);
    const float x = dt_bias[h * DD + tid];
    const float sp = (x > 20.f) ? x : log1pf(__expf(x));   // softplus
    wdec[tid] = aexp * sp;
  }
  __syncthreads();
  for (int i = tid; i < (CH + 1) * DD; i += 256) {
    const int e = i >> 7, d = i & (DD - 1);
    lampow[e][d] = __expf(-wdec[d] * (float)e);  // lambda^e (only decays)
  }
  for (int i = tid; i < DD * DD; i += 256) ((float*)S)[i] = 0.f;
  __syncthreads();

  const int pr  = tid >> 3;                      // phase-1 row 0..31
  const int pd0 = (tid & 7) * 16;                // 16 contiguous channels

  for (int c0 = 0; c0 < TT; c0 += CH) {
    // ---- phase 1: load chunk, stage scaled bf16 operand tiles ----
    {
      const size_t grow = ((size_t)(b * TT + c0 + pr) * HH + h) * DD;
      const float bet = sigmoidf_(beta_logit[(size_t)(b * TT + c0 + pr) * HH + h]);
      #pragma unroll
      for (int j = 0; j < 16; ++j) {
        const int d = pd0 + j;
        const float qv = bf2f(q[grow + d]);
        const float kv = bf2f(k[grow + d]);
        const float vv = bf2f(v[grow + d]);
        qb[pr][d]  = f2bf(qv);
        kb_[pr][d] = f2bf(kv);
        qsb[pr][d] = f2bf(qv * lampow[pr + 1][d]);
        kdbT[d][pr] = f2bf(kv * lampow[CH - 1 - pr][d]);
        vbbT[d][pr] = f2bf(bet * vv);
        if (pr >= 16) qlb[pr - 16][d] = f2bf(qv * lampow[pr - 16][d]);
        else          klb[pr][d]      = f2bf(kv * lampow[16 - pr][d]);
      }
    }
    __syncthreads();
    // ---- phase 2: intra-chunk attention A[r][s] = sum_d q_r k_s lam^{r-s} ----
    Ab[tid >> 4][16 + (tid & 15)] = 0;           // upper-right block = 0 (masked)
    for (int p = tid; p < 512; p += 256) {       // two diagonal 16x16 blocks, direct
      const int blk = p >> 8, rl = (p >> 4) & 15, sl = p & 15;
      const int r = blk * 16 + rl, s = blk * 16 + sl;
      float a = 0.f;
      if (sl <= rl) {
        const float* lp = &lampow[rl - sl][0];
        for (int d = 0; d < DD; ++d)
          a += bf2f(qb[r][d]) * bf2f(kb_[s][d]) * lp[d];
      }
      Ab[r][s] = f2bf(a);
    }
    if (wave == 0) {                             // lower-left block: WMMA, safe scaling
      v8f cacc = zero8();
      #pragma unroll
      for (int kt = 0; kt < 4; ++kt) {
        v16bf af = ldfrag(&qlb[lo][kt * 32 + hi * 8], &qlb[lo][kt * 32 + 16 + hi * 8]);
        v16bf bf = ldfrag(&klb[lo][kt * 32 + hi * 8], &klb[lo][kt * 32 + 16 + hi * 8]);
        cacc = wmma_bf16(af, bf, cacc);
      }
      #pragma unroll
      for (int j = 0; j < 8; ++j) Ab[16 + hi * 8 + j][lo] = f2bf(cacc[j]);
    }
    __syncthreads();
    // ---- phase 3: O = Qs@S + A@Vb ; S = diag(lam^32)S + Kd^T@Vb ----
    const int dv = wave * 16 + lo;               // wave-exclusive value column
    #pragma unroll
    for (int mt = 0; mt < 2; ++mt) {
      v8f cacc = zero8();
      const int qr = mt * 16 + lo;
      #pragma unroll
      for (int kt = 0; kt < 4; ++kt) {           // inter-chunk: Qs @ S_old
        v16bf af = ldfrag(&qsb[qr][kt * 32 + hi * 16], &qsb[qr][kt * 32 + hi * 16 + 8]);
        FragU bf;
        #pragma unroll
        for (int e = 0; e < 16; ++e) bf.s[e] = f2bf(S[kt * 32 + hi * 16 + e][dv]);
        cacc = wmma_bf16(af, bf.v, cacc);
      }
      v16bf aA = ldfrag(&Ab[qr][hi * 8], &Ab[qr][16 + hi * 8]);          // intra: A @ Vb
      v16bf bV = ldfrag(&vbbT[dv][hi * 8], &vbbT[dv][16 + hi * 8]);
      cacc = wmma_bf16(aA, bV, cacc);
      #pragma unroll
      for (int j = 0; j < 8; ++j) {
        const int r = mt * 16 + hi * 8 + j;
        o[((size_t)(b * TT + c0 + r) * HH + h) * DD + dv] = cacc[j];
      }
    }
    {                                            // state update, rank-32
      v16bf bV = ldfrag(&vbbT[dv][hi * 8], &vbbT[dv][16 + hi * 8]);
      #pragma unroll
      for (int mt = 0; mt < 8; ++mt) {
        v16bf aK = ldfrag(&kdbT[mt * 16 + lo][hi * 8], &kdbT[mt * 16 + lo][16 + hi * 8]);
        v8f dS = wmma_bf16(aK, bV, zero8());
        #pragma unroll
        for (int j = 0; j < 8; ++j) {
          const int dk = mt * 16 + hi * 8 + j;
          S[dk][dv] = lampow[CH][dk] * S[dk][dv] + dS[j];
        }
      }
    }
    __syncthreads();
  }
}

// ---------------------------------------------------------------------------
// FusedRMSNormGated: o = o * rsqrt(mean(o^2)+eps) * w * sigmoid(g), per head.
// Also emits the bf16 A-operand copy for the final output GEMM.
// ---------------------------------------------------------------------------
__global__ __launch_bounds__(256) void gate_norm_kernel(
    float* __restrict__ o, const unsigned short* __restrict__ g,
    const float* __restrict__ norm_w, unsigned short* __restrict__ obf)
{
  const int tid  = threadIdx.x;
  const int wave = tid >> 5, lane = tid & 31;
  const int gi = blockIdx.x * 8 + wave;
  const int h = gi & (HH - 1);
  const int t = (gi >> 4) & (TT - 1);
  const int b = gi >> 15;
  const int d = lane * 4;
  const size_t base = ((size_t)(b * TT + t) * HH + h) * DD + d;
  float ov[4];
  #pragma unroll
  for (int i = 0; i < 4; ++i) ov[i] = o[base + i];
  float ss = ov[0]*ov[0] + ov[1]*ov[1] + ov[2]*ov[2] + ov[3]*ov[3];
  for (int m = 16; m >= 1; m >>= 1) ss += __shfl_xor(ss, m, 32);
  const float r = rsqrtf(ss * (1.f / DD) + 1e-5f);
  #pragma unroll
  for (int i = 0; i < 4; ++i) {
    const float val = ov[i] * r * norm_w[d + i] * sigmoidf_(bf2f(g[base + i]));
    o[base + i]   = val;
    obf[base + i] = f2bf(val);
  }
}

// ---------------------------------------------------------------------------
extern "C" void kernel_launch(void* const* d_in, const int* in_sizes, int n_in,
                              void* d_out, int out_size, void* d_ws, size_t ws_size,
                              hipStream_t stream)
{
  const float* x     = (const float*)d_in[0];
  const float* Wq    = (const float*)d_in[1];
  const float* Wk    = (const float*)d_in[2];
  const float* Wv    = (const float*)d_in[3];
  const float* wq_c  = (const float*)d_in[4];
  const float* wk_c  = (const float*)d_in[5];
  const float* wv_c  = (const float*)d_in[6];
  const float* Wb    = (const float*)d_in[7];
  const float* A_log = (const float*)d_in[8];
  const float* dtb   = (const float*)d_in[9];
  const float* Wg1   = (const float*)d_in[10];
  const float* Wg2   = (const float*)d_in[11];
  const float* bg2   = (const float*)d_in[12];
  const float* normw = (const float*)d_in[13];
  const float* Wo    = (const float*)d_in[14];

  const size_t NT   = (size_t)BB * TT;       // 4096 token rows
  const size_t NTKD = NT * KDIM;             // 8,388,608 elems
  const size_t WW   = (size_t)HIDc * KDIM;   // 4,194,304 elems

  // Workspace layout (256B-aligned regions)
  char* base = (char*)d_ws;
  size_t off = 0;
  auto alloc = [&](size_t bytes) -> void* {
    off = (off + 255) & ~(size_t)255;
    void* p = base + off;
    off += bytes;
    return p;
  };
  unsigned short* xb    = (unsigned short*)alloc(NTKD * 2);        // bf16(x)
  unsigned short* WqT   = (unsigned short*)alloc(WW * 2);
  unsigned short* WkT   = (unsigned short*)alloc(WW * 2);
  unsigned short* WvT   = (unsigned short*)alloc(WW * 2);
  unsigned short* WoT   = (unsigned short*)alloc(WW * 2);
  unsigned short* WbT   = (unsigned short*)alloc((size_t)HH * HIDc * 2);
  unsigned short* Wg1T  = (unsigned short*)alloc((size_t)DD * HIDc * 2);
  unsigned short* Wg2T  = (unsigned short*)alloc((size_t)KDIM * DD * 2);
  unsigned short* qpre  = (unsigned short*)alloc(NTKD * 2);
  unsigned short* kpre  = (unsigned short*)alloc(NTKD * 2);
  unsigned short* vpre  = (unsigned short*)alloc(NTKD * 2);
  unsigned short* qn    = (unsigned short*)alloc(NTKD * 2);
  unsigned short* kn    = (unsigned short*)alloc(NTKD * 2);
  unsigned short* vn    = (unsigned short*)alloc(NTKD * 2);
  unsigned short* g1bf  = (unsigned short*)alloc(NT * DD * 2);
  unsigned short* gbuf  = (unsigned short*)alloc(NTKD * 2);
  unsigned short* obf   = (unsigned short*)alloc(NTKD * 2);
  float*          obuf  = (float*)alloc(NTKD * 4);
  float*          betal = (float*)alloc(NT * HH * 4);
  (void)ws_size;

  dim3 blk(256);
  dim3 gFull((KDIM + 127) / 128, (int)((NT + 127) / 128));   // 16 x 32
  dim3 gNarrow(1, (int)((NT + 127) / 128));                  // N <= 128

  // --- one-time operand preparation (bf16 pack / transpose) ---
  pack_bf16_kernel<<<4096, blk, 0, stream>>>(x, xb, (int)NTKD);
  transpose_bf16_kernel<<<dim3(KDIM / 32, HIDc / 32), blk, 0, stream>>>(Wq,  WqT,  HIDc, KDIM);
  transpose_bf16_kernel<<<dim3(KDIM / 32, HIDc / 32), blk, 0, stream>>>(Wk,  WkT,  HIDc, KDIM);
  transpose_bf16_kernel<<<dim3(KDIM / 32, HIDc / 32), blk, 0, stream>>>(Wv,  WvT,  HIDc, KDIM);
  transpose_bf16_kernel<<<dim3(HIDc / 32, KDIM / 32), blk, 0, stream>>>(Wo,  WoT,  KDIM, HIDc);
  transpose_bf16_kernel<<<dim3(1,         HIDc / 32), blk, 0, stream>>>(Wb,  WbT,  HIDc, HH);
  transpose_bf16_kernel<<<dim3(DD / 32,   HIDc / 32), blk, 0, stream>>>(Wg1, Wg1T, HIDc, DD);
  transpose_bf16_kernel<<<dim3(KDIM / 32, DD / 32),   blk, 0, stream>>>(Wg2, Wg2T, DD, KDIM);

  // --- projections (compute-bound bf16 WMMA GEMMs, async double-buffered) ---
  gemm_bf16_kernel<<<gFull,   blk, 0, stream>>>(xb, WqT,  nullptr, nullptr, qpre, (int)NT, KDIM, HIDc);
  gemm_bf16_kernel<<<gFull,   blk, 0, stream>>>(xb, WkT,  nullptr, nullptr, kpre, (int)NT, KDIM, HIDc);
  gemm_bf16_kernel<<<gFull,   blk, 0, stream>>>(xb, WvT,  nullptr, nullptr, vpre, (int)NT, KDIM, HIDc);
  gemm_bf16_kernel<<<gNarrow, blk, 0, stream>>>(xb, WbT,  nullptr, betal, nullptr, (int)NT, HH, HIDc);
  gemm_bf16_kernel<<<gNarrow, blk, 0, stream>>>(xb, Wg1T, nullptr, nullptr, g1bf, (int)NT, DD, HIDc);
  gemm_bf16_kernel<<<gFull,   blk, 0, stream>>>(g1bf, Wg2T, bg2,   nullptr, gbuf, (int)NT, KDIM, DD);

  // --- short conv + SiLU (+ l2norm on q,k), bf16 streams ---
  const int convBlocks = (BB * TT * HH) / 8;   // one wave per (b,t,h)
  conv_silu_kernel<<<convBlocks, blk, 0, stream>>>(qpre, wq_c, qn, 1);
  conv_silu_kernel<<<convBlocks, blk, 0, stream>>>(kpre, wk_c, kn, 1);
  conv_silu_kernel<<<convBlocks, blk, 0, stream>>>(vpre, wv_c, vn, 0);

  // --- chunked decay scan (one WGP-resident workgroup per (b,h)) ---
  quasar_scan_kernel<<<BB * HH, blk, 0, stream>>>(qn, kn, vn, betal, A_log, dtb, obuf);

  // --- gated RMSNorm (emits bf16 copy), then output projection ---
  gate_norm_kernel<<<convBlocks, blk, 0, stream>>>(obuf, gbuf, normw, obf);
  gemm_bf16_kernel<<<gFull, blk, 0, stream>>>(obf, WoT, nullptr, (float*)d_out, nullptr,
                                              (int)NT, HIDc, KDIM);
}
